// Longhorn_8358006358686
// MI455X (gfx1250) — compile-verified, matrix operand order checked by reading, not run on hardware
//
#include <hip/hip_runtime.h>
#include <hip/hip_bf16.h>
#include <stdint.h>

// ---------------- model constants ----------------
#define BATCH   2
#define SEQL    2048
#define DMODEL  1024
#define DINNER  2048
#define DTRANK  64
#define DSTATE  16
#define DCONV   4
#define MROWS   (BATCH * SEQL)        // 4096
#define XDBLW   (DTRANK + 2 * DSTATE) // 96

// ---------------- WMMA types ----------------
typedef __attribute__((ext_vector_type(16))) __bf16 v16bf;
typedef __attribute__((ext_vector_type(8)))  __bf16 v8bf;
typedef __attribute__((ext_vector_type(8)))  float  v8f;

#define LDSTRIDE 40   // halves per LDS tile row: 80B = 5*16B (b128-aligned, bank-conflict-free)

__device__ __forceinline__ unsigned pack_bf16x2(float a, float b) {
    unsigned ua = __builtin_bit_cast(unsigned, a);
    unsigned ub = __builtin_bit_cast(unsigned, b);
    return (ua >> 16) | (ub & 0xFFFF0000u);      // truncating f32->bf16, packed
}
__device__ __forceinline__ __bf16 f2bf(float f) {
    unsigned u = __builtin_bit_cast(unsigned, f);
    return __builtin_bit_cast(__bf16, (unsigned short)(u >> 16));
}

__device__ __forceinline__ void async_load16(uint32_t lds_byte, const __bf16* gaddr) {
    // CDNA5 async global->LDS copy, 16B per lane, tracked by ASYNCcnt
    asm volatile("global_load_async_to_lds_b128 %0, %1, off"
                 :: "v"(lds_byte), "v"(gaddr) : "memory");
}
__device__ __forceinline__ void wait_async(bool keep4) {
    if (keep4) asm volatile("s_wait_asynccnt 4" ::: "memory");
    else       asm volatile("s_wait_asynccnt 0" ::: "memory");
}

__device__ __forceinline__ float sigmoidf_fast(float x) {
    return 1.0f / (1.0f + __expf(-x));
}

// =====================================================================
// fp32 -> bf16 pack kernel (4 elements / thread, memory bound)
// =====================================================================
__global__ __launch_bounds__(256) void cvt_bf16_kernel(
    const float* __restrict__ src, unsigned* __restrict__ dst /* bf16x2 */)
{
    int i = blockIdx.x * 256 + threadIdx.x;       // over n/4
    float4 v = ((const float4*)src)[i];
    uint2 p;
    p.x = pack_bf16x2(v.x, v.y);
    p.y = pack_bf16x2(v.z, v.w);
    ((uint2*)dst)[i] = p;
}

// =====================================================================
// Generic GEMM: C[M,N] = A[M,K] * W[N,K]^T   (bf16 in, fp32 out)
// Block tile 128(M) x 128(N), K-step 32. 8 waves: 2 (M) x 4 (N), each
// computing a 64x32 tile = 4x2 fragments = 8 WMMAs per K-step against
// 12 ds_load_b128 (~21 FLOP / LDS byte). bf16 tiles staged with async
// global->LDS, double-buffered (40 KB LDS).
// =====================================================================
__global__ __launch_bounds__(256) void gemm_bf16_wmma(
    const __bf16* __restrict__ A, int lda,
    const __bf16* __restrict__ W, int ldw,
    float* __restrict__ C, int ldc,
    int M, int N, int K)
{
    __shared__ __attribute__((aligned(16))) __bf16 As[2][128 * LDSTRIDE];
    __shared__ __attribute__((aligned(16))) __bf16 Bs[2][128 * LDSTRIDE];

    const int tid   = threadIdx.x;
    const int lane  = tid & 31;
    const int wave  = tid >> 5;
    const int waveM = (wave >> 2) * 64;   // 0 or 64
    const int waveN = (wave &  3) * 32;   // 0,32,64,96
    const int gm0   = blockIdx.y * 128;
    const int gn0   = blockIdx.x * 128;
    const int kt    = K >> 5;             // K is a multiple of 32 at every call site

    auto issue = [&](int s, int kofs) {
        // A tile: 128 rows x 32 halves = 512 x 16B chunks -> 2 per thread
        #pragma unroll
        for (int i = 0; i < 2; ++i) {
            int c = tid + 256 * i;
            int row = c >> 2, ch = c & 3;         // chunk = 8 halves
            int gr = gm0 + row; if (gr >= M) gr = M - 1;
            uint32_t dst = (uint32_t)(uintptr_t)&As[s][row * LDSTRIDE + ch * 8];
            async_load16(dst, A + (size_t)gr * lda + kofs + ch * 8);
        }
        // B tile: 128 rows x 32 halves -> 2 per thread
        #pragma unroll
        for (int i = 0; i < 2; ++i) {
            int c = tid + 256 * i;
            int row = c >> 2, ch = c & 3;
            int gr = gn0 + row; if (gr >= N) gr = N - 1;   // clamp (N=96 tile)
            uint32_t dst = (uint32_t)(uintptr_t)&Bs[s][row * LDSTRIDE + ch * 8];
            async_load16(dst, W + (size_t)gr * ldw + kofs + ch * 8);
        }
    };

    v8f acc[4][2] = {};
    int cur = 0;
    issue(0, 0);

    for (int kk = 0; kk < kt; ++kk) {
        const bool more = (kk + 1 < kt);
        if (more) issue(cur ^ 1, (kk + 1) << 5);
        wait_async(more);
        __syncthreads();

        const int khalf = (lane < 16) ? 0 : 8;  // 16-bit fragment lane split

        v16bf afrag[4], bfrag[2];
        #pragma unroll
        for (int tm = 0; tm < 4; ++tm) {
            const __bf16* ar = &As[cur][(waveM + tm * 16 + (lane & 15)) * LDSTRIDE];
            v8bf lo = *(const v8bf*)(ar + khalf);        // K = khalf..khalf+7
            v8bf hi = *(const v8bf*)(ar + 16 + khalf);   // K = 16+khalf..+7
            afrag[tm] = __builtin_shufflevector(lo, hi, 0,1,2,3,4,5,6,7,8,9,10,11,12,13,14,15);
        }
        #pragma unroll
        for (int tn = 0; tn < 2; ++tn) {
            const __bf16* br = &Bs[cur][(waveN + tn * 16 + (lane & 15)) * LDSTRIDE];
            v8bf lo = *(const v8bf*)(br + khalf);
            v8bf hi = *(const v8bf*)(br + 16 + khalf);
            bfrag[tn] = __builtin_shufflevector(lo, hi, 0,1,2,3,4,5,6,7,8,9,10,11,12,13,14,15);
        }

        #pragma unroll
        for (int tm = 0; tm < 4; ++tm)
            #pragma unroll
            for (int tn = 0; tn < 2; ++tn)
                acc[tm][tn] = __builtin_amdgcn_wmma_f32_16x16x32_bf16(
                    false, afrag[tm], false, bfrag[tn],
                    (short)0, acc[tm][tn], false, false);

        __syncthreads();
        cur ^= 1;
    }

    // epilogue: VGPR r holds M=r (lanes 0-15) / M=8+r (lanes 16-31)
    #pragma unroll
    for (int tm = 0; tm < 4; ++tm)
        #pragma unroll
        for (int tn = 0; tn < 2; ++tn) {
            const int ncol  = gn0 + waveN + tn * 16 + (lane & 15);
            const int mbase = gm0 + waveM + tm * 16 + ((lane < 16) ? 0 : 8);
            if (ncol < N) {
                #pragma unroll
                for (int r = 0; r < 8; ++r)
                    C[(size_t)(mbase + r) * ldc + ncol] = acc[tm][tn][r];
            }
        }
}

// =====================================================================
// Causal depthwise conv1d (d_conv=4) + bias + SiLU.
// Dual-writes fp32 (for the scan) and bf16 (for the x_proj GEMM).
// =====================================================================
__global__ __launch_bounds__(256) void conv_silu_kernel(
    const float* __restrict__ XZ, const float* __restrict__ cw,
    const float* __restrict__ cb, float* __restrict__ XC,
    __bf16* __restrict__ XCbf)
{
    int idx = blockIdx.x * 256 + threadIdx.x;           // over MROWS*DINNER
    int d   = idx & (DINNER - 1);
    int row = idx >> 11;                                // /DINNER
    int l   = row & (SEQL - 1);
    int b   = row >> 11;                                // /SEQL
    float acc = cb[d];
    #pragma unroll
    for (int i = 0; i < DCONV; ++i) {
        int li = l - (DCONV - 1) + i;
        if (li >= 0)
            acc += XZ[(size_t)(b * SEQL + li) * (2 * DINNER) + d] * cw[d * DCONV + i];
    }
    float s = acc * sigmoidf_fast(acc);                 // SiLU
    XC[(size_t)row * DINNER + d]   = s;
    XCbf[(size_t)row * DINNER + d] = f2bf(s);
}

// =====================================================================
// Selective scan: one thread per (b, channel), 16 states in registers,
// k/q broadcast via LDS. Writes gated output directly as bf16 for the
// final out_proj GEMM.
// =====================================================================
__global__ __launch_bounds__(256) void scan_kernel(
    const float* __restrict__ XC, const float* __restrict__ DT,
    const float* __restrict__ XDBL, const float* __restrict__ XZ,
    const float* __restrict__ dt_b, const float* __restrict__ Dv,
    __bf16* __restrict__ Ybf)
{
    const int b  = blockIdx.x >> 3;
    const int d  = ((blockIdx.x & 7) << 8) + threadIdx.x;
    const float bias = dt_b[d];
    const float Dd   = Dv[d];

    __shared__ float kq[2 * DSTATE];
    float st[DSTATE];
    #pragma unroll
    for (int n = 0; n < DSTATE; ++n) st[n] = 0.0f;

    for (int t = 0; t < SEQL; ++t) {
        const size_t row = (size_t)(b * SEQL + t);
        if (threadIdx.x < 2 * DSTATE)
            kq[threadIdx.x] = XDBL[row * XDBLW + DTRANK + threadIdx.x];
        __syncthreads();

        float ksum = 0.0f;
        #pragma unroll
        for (int n = 0; n < DSTATE; ++n) ksum += kq[n] * kq[n];

        const float x  = XC[row * DINNER + d];
        const float dr = DT[row * DINNER + d];
        const float z  = XZ[row * (2 * DINNER) + DINNER + d];

        float dtv = sigmoidf_fast(dr + bias);
        dtv = dtv / (1.0f + dtv * ksum);

        float y = 0.0f;
        #pragma unroll
        for (int n = 0; n < DSTATE; ++n) {
            const float kn = kq[n];
            st[n] = st[n] * (1.0f - dtv * kn * kn) + x * dtv * kn;
            y += st[n] * kq[DSTATE + n];
        }
        Ybf[row * DINNER + d] = f2bf((y + Dd * x) * z * sigmoidf_fast(z));
        __syncthreads();
    }
}

// =====================================================================
// Host-side orchestration
// =====================================================================
extern "C" void kernel_launch(void* const* d_in, const int* in_sizes, int n_in,
                              void* d_out, int out_size, void* d_ws, size_t ws_size,
                              hipStream_t stream) {
    const float* hs    = (const float*)d_in[0];  // [B,L,DMODEL]
    const float* inW   = (const float*)d_in[1];  // [2*DINNER, DMODEL]
    const float* convW = (const float*)d_in[2];  // [DINNER,1,DCONV]
    const float* convB = (const float*)d_in[3];  // [DINNER]
    const float* xpW   = (const float*)d_in[4];  // [XDBLW, DINNER]
    const float* dthW  = (const float*)d_in[5];  // [DINNER, DTRANK]
    const float* dthB  = (const float*)d_in[6];  // [DINNER]
    const float* outW  = (const float*)d_in[7];  // [DMODEL, DINNER]
    const float* Dv    = (const float*)d_in[8];  // [DINNER]
    float* out = (float*)d_out;                  // [B,L,DMODEL]

    // ---- workspace layout ----
    float* ws   = (float*)d_ws;
    float* XZ   = ws;                                   // MROWS x 4096 fp32
    float* XC   = XZ   + (size_t)MROWS * (2 * DINNER);  // MROWS x 2048 fp32
    float* XDBL = XC   + (size_t)MROWS * DINNER;        // MROWS x 96   fp32
    float* DT   = XDBL + (size_t)MROWS * XDBLW;         // MROWS x 2048 fp32
    __bf16* bfb   = (__bf16*)(DT + (size_t)MROWS * DINNER);
    __bf16* hs_b  = bfb;                                   // MROWS x 1024
    __bf16* inW_b = hs_b  + (size_t)MROWS * DMODEL;        // 4096 x 1024
    __bf16* XC_b  = inW_b + (size_t)(2*DINNER) * DMODEL;   // MROWS x 2048
    __bf16* xpW_b = XC_b  + (size_t)MROWS * DINNER;        // 96 x 2048
    __bf16* XDBL_b= xpW_b + (size_t)XDBLW * DINNER;        // MROWS x 96
    __bf16* dthW_b= XDBL_b+ (size_t)MROWS * XDBLW;         // 2048 x 64
    __bf16* Y_b   = dthW_b+ (size_t)DINNER * DTRANK;       // MROWS x 2048
    __bf16* outW_b= Y_b   + (size_t)MROWS * DINNER;        // 1024 x 2048

    dim3 blk(256);
    #define CVT(src, dst, n) \
        cvt_bf16_kernel<<<(n) / 1024, blk, 0, stream>>>((src), (unsigned*)(dst))

    // ---- one-time bf16 packs (memory bound, ~5us total at 23.3 TB/s) ----
    CVT(hs,   hs_b,   (size_t)MROWS * DMODEL);
    CVT(inW,  inW_b,  (size_t)(2*DINNER) * DMODEL);
    CVT(xpW,  xpW_b,  (size_t)XDBLW * DINNER);
    CVT(dthW, dthW_b, (size_t)DINNER * DTRANK);
    CVT(outW, outW_b, (size_t)DMODEL * DINNER);

    // 1) in_proj: XZ = hs @ inW^T     (M=4096, N=4096, K=1024)
    gemm_bf16_wmma<<<dim3((2*DINNER)/128, MROWS/128), blk, 0, stream>>>(
        hs_b, DMODEL, inW_b, DMODEL, XZ, 2*DINNER, MROWS, 2*DINNER, DMODEL);

    // 2) causal depthwise conv + SiLU -> XC (fp32) + XC_b (bf16)
    conv_silu_kernel<<<(MROWS * DINNER) / 256, blk, 0, stream>>>(
        XZ, convW, convB, XC, XC_b);

    // 3) x_proj: XDBL = XC @ xpW^T    (M=4096, N=96, K=2048)
    gemm_bf16_wmma<<<dim3(1, MROWS/128), blk, 0, stream>>>(
        XC_b, DINNER, xpW_b, DINNER, XDBL, XDBLW, MROWS, XDBLW, DINNER);
    CVT(XDBL, XDBL_b, (size_t)MROWS * XDBLW);

    // 4) dt_head: DT = XDBL[:, :64] @ dthW^T   (M=4096, N=2048, K=64)
    gemm_bf16_wmma<<<dim3(DINNER/128, MROWS/128), blk, 0, stream>>>(
        XDBL_b, XDBLW, dthW_b, DTRANK, DT, DINNER, MROWS, DINNER, DTRANK);

    // 5) selective scan + skip + gating -> Y (bf16)
    scan_kernel<<<(BATCH * DINNER) / 256, blk, 0, stream>>>(
        XC, DT, XDBL, XZ, dthB, Dv, Y_b);

    // 6) out_proj: out = Y @ outW^T   (M=4096, N=1024, K=2048)
    gemm_bf16_wmma<<<dim3(DMODEL/128, MROWS/128), blk, 0, stream>>>(
        Y_b, DINNER, outW_b, DINNER, out, DMODEL, MROWS, DMODEL, DINNER);
    #undef CVT
}
